// ImprovedGeometricDeepONet_9818295239421
// MI455X (gfx1250) — matrix operand order repaired and checked by reading, not compiled, not run on hardware
//
#include <hip/hip_runtime.h>
#include <cmath>

typedef _Float16 half_t;
typedef __attribute__((ext_vector_type(16))) _Float16 v16h;
typedef __attribute__((ext_vector_type(8)))  _Float16 v8h;
typedef __attribute__((ext_vector_type(8)))  float    v8f;

#define HID   128
#define NREFS 10
#define NC    36
#define PDIM  64
#define NPB   8192      // points per batch
#define MT    128       // rows per trunk block
#define BATCH 64
#define PI_F  3.14159265358979323846f

// padded leading dims (halves) -> conflict-free LDS banks, 16B-aligned chunks
#define LDA   40        // features / W0 row stride (K padded 32 -> 40)
#define LDH   136       // h / W1 row stride (K=128 -> 136)

// ws layout (bytes)
#define WS_HW0   0                       // 128*40 halves  = 10240 B
#define WS_HW1   10240                   // 128*136 halves = 34816 B
#define WS_WCOMB 45056                   // 64*128 f32     = 32768 B
#define WS_CC    77824                   // 64 f32

static __device__ __forceinline__ v8f wmma_f16(v16h a, v16h b, v8f c) {
  return __builtin_amdgcn_wmma_f32_16x16x32_f16(
      false, a, false, b, (short)0, c, false, false);
}

static __device__ __forceinline__ v16h cat8(v8h lo, v8h hi) {
  return __builtin_shufflevector(lo, hi, 0,1,2,3,4,5,6,7,8,9,10,11,12,13,14,15);
}

// A-fragment (16xK tile of W, row-major, ldk halves): lane m = ln,
// K runs [kb+lh*8 .. +7] and [kb+16+lh*8 .. +7]
static __device__ __forceinline__ v16h ld_afrag(const _Float16* W, int row,
                                                int ldk, int kb, int lh) {
  const _Float16* p = W + row * ldk + kb + lh * 8;
  v8h lo = *(const v8h*)(p);
  v8h hi = *(const v8h*)(p + 16);
  return cat8(lo, hi);
}

// B-fragment (Kx16 tile of activations, row-major [row][k], ldk halves):
// lane col n = ln -> K runs [kb+lh*8 .. +7] and [kb+16+lh*8 .. +7]
static __device__ __forceinline__ v16h ld_bfrag(const _Float16* H, int row,
                                                int ldk, int kb, int lh) {
  const _Float16* p = H + row * ldk + kb + lh * 8;
  v8h lo = *(const v8h*)(p);
  v8h hi = *(const v8h*)(p + 16);
  return cat8(lo, hi);
}

// ============ prep kernel: weights -> padded f16 images in ws ============
__global__ __launch_bounds__(256) void prep_kernel(
    const float* __restrict__ tw0, const float* __restrict__ tb0,
    const float* __restrict__ tw1, _Float16* __restrict__ hW0,
    _Float16* __restrict__ hW1)
{
  const int NW0 = HID * LDA;     // 5120
  const int NW1 = HID * LDH;     // 17408
  for (int idx = blockIdx.x * blockDim.x + threadIdx.x; idx < NW0 + NW1;
       idx += gridDim.x * blockDim.x) {
    if (idx < NW0) {
      int n = idx / LDA, k = idx % LDA;
      float v = (k < 13) ? tw0[n * 13 + k] : (k == 13 ? tb0[n] : 0.0f);
      hW0[idx] = (_Float16)v;
    } else {
      int j = idx - NW0;
      int n = j / LDH, k = j % LDH;
      hW1[j] = (_Float16)((k < HID) ? tw1[n * HID + k] : 0.0f);
    }
  }
}

// =====================  branch kernel (tiny)  =====================
__global__ __launch_bounds__(128) void branch_kernel(
    const float* __restrict__ coeffs, const float* __restrict__ cmean,
    const float* __restrict__ cstd,
    const float* __restrict__ w0, const float* __restrict__ b0,
    const float* __restrict__ g0, const float* __restrict__ be0,
    const float* __restrict__ w1, const float* __restrict__ b1,
    const float* __restrict__ g1, const float* __restrict__ be1,
    const float* __restrict__ w2, const float* __restrict__ b2,
    const float* __restrict__ g2, const float* __restrict__ be2,
    const float* __restrict__ bwo, const float* __restrict__ bbo,
    const float* __restrict__ two, const float* __restrict__ tbo,
    float* __restrict__ wcomb, float* __restrict__ cconst)
{
  const int b = blockIdx.x;
  const int t = threadIdx.x;
  __shared__ float sc[NC];
  __shared__ float sx[HID];
  __shared__ float s1[HID], s2[HID];
  __shared__ float spb[PDIM];

  if (t < NC) sc[t] = (coeffs[b * NC + t] - cmean[t]) / (cstd[t] + 1e-8f);
  __syncthreads();

  float v = b0[t];
  for (int k = 0; k < NC; ++k) v += w0[t * NC + k] * sc[k];
  s1[t] = v; s2[t] = v * v; __syncthreads();
  for (int off = 64; off > 0; off >>= 1) {
    if (t < off) { s1[t] += s1[t + off]; s2[t] += s2[t + off]; }
    __syncthreads();
  }
  float mu = s1[0] * (1.0f / HID);
  float var = s2[0] * (1.0f / HID) - mu * mu;
  float x = fmaxf((v - mu) * rsqrtf(var + 1e-5f) * g0[t] + be0[t], 0.0f);
  sx[t] = x; __syncthreads();

  v = b1[t];
  for (int k = 0; k < HID; ++k) v += w1[t * HID + k] * sx[k];
  s1[t] = v; s2[t] = v * v; __syncthreads();
  for (int off = 64; off > 0; off >>= 1) {
    if (t < off) { s1[t] += s1[t + off]; s2[t] += s2[t + off]; }
    __syncthreads();
  }
  mu = s1[0] * (1.0f / HID);
  var = s2[0] * (1.0f / HID) - mu * mu;
  x = fmaxf((v - mu) * rsqrtf(var + 1e-5f) * g1[t] + be1[t], 0.0f) + sx[t];
  __syncthreads();
  sx[t] = x; __syncthreads();

  v = b2[t];
  for (int k = 0; k < HID; ++k) v += w2[t * HID + k] * sx[k];
  s1[t] = v; s2[t] = v * v; __syncthreads();
  for (int off = 64; off > 0; off >>= 1) {
    if (t < off) { s1[t] += s1[t + off]; s2[t] += s2[t + off]; }
    __syncthreads();
  }
  mu = s1[0] * (1.0f / HID);
  var = s2[0] * (1.0f / HID) - mu * mu;
  x = fmaxf((v - mu) * rsqrtf(var + 1e-5f) * g2[t] + be2[t], 0.0f) + sx[t];
  __syncthreads();
  sx[t] = x; __syncthreads();

  if (t < PDIM) {
    float pv = bbo[t];
    for (int k = 0; k < HID; ++k) pv += bwo[t * HID + k] * sx[k];
    spb[t] = pv;
  }
  __syncthreads();

  float wv = 0.0f;
  for (int p = 0; p < PDIM; ++p) wv += spb[p] * two[p * HID + t];
  wcomb[b * HID + t] = wv;
  if (t == 0) {
    float cc = 0.0f;
    for (int p = 0; p < PDIM; ++p) cc += spb[p] * tbo[p];
    cconst[b] = cc;
  }
}

// =====================  trunk kernel (WMMA)  =====================
// D = W(16 x K) * Act(K x 16rows). Lane holds row n=ln, neurons m=mt*16+lh*8+i.
__global__ __launch_bounds__(256) void trunk_kernel(
    const float* __restrict__ coords,
    const float* __restrict__ ref_theta, const float* __restrict__ ref_phi,
    const _Float16* __restrict__ hW0, const _Float16* __restrict__ hW1,
    const float* __restrict__ tg0, const float* __restrict__ tbe0,
    const float* __restrict__ tb1, const float* __restrict__ tg1,
    const float* __restrict__ tbe1,
    const float* __restrict__ wcomb, const float* __restrict__ cconst,
    float* __restrict__ out)
{
  __shared__ alignas(16) _Float16 sW0[HID * LDA];   // [n_out][k], 10240 B
  __shared__ alignas(16) _Float16 sW1[HID * LDH];   // [n_out][k], 34816 B
  __shared__ alignas(16) _Float16 sA [MT * LDA];    // [row][k],   10240 B
  __shared__ alignas(16) _Float16 sH [MT * LDH];    // [row][k],   34816 B
  __shared__ alignas(32) float sG0[HID], sBe0[HID];
  __shared__ alignas(32) float sB1[HID], sG1[HID], sBe1[HID];
  __shared__ alignas(32) float sWc[HID];
  __shared__ float sRct[NREFS], sRst[NREFS], sRp[NREFS];

  const int tid  = threadIdx.x;
  const int wave = tid >> 5;
  const int lane = tid & 31;
  const int ln   = lane & 15;
  const int lh   = lane >> 4;

  const int row0 = blockIdx.x * MT;
  const int b    = row0 / NPB;

  // ---------------- phase 1: stage weights (pure 16B copies) ----------------
  {
    const uint4* srcW0 = (const uint4*)hW0;
    const uint4* srcW1 = (const uint4*)hW1;
    uint4* dstW0 = (uint4*)sW0;
    uint4* dstW1 = (uint4*)sW1;
    const int NQ0 = HID * LDA / 8;    // 640 uint4
    const int NQ1 = HID * LDH / 8;    // 2176 uint4
    for (int i = tid; i < NQ0; i += 256) dstW0[i] = srcW0[i];
    for (int i = tid; i < NQ1; i += 256) dstW1[i] = srcW1[i];
  }
  if (tid < HID) {
    sG0[tid] = tg0[tid];  sBe0[tid] = tbe0[tid];
    sB1[tid] = tb1[tid];  sG1[tid]  = tg1[tid];  sBe1[tid] = tbe1[tid];
    sWc[tid] = wcomb[b * HID + tid];
  }
  if (tid < NREFS) {
    float rt = ref_theta[tid];
    sRct[tid] = cosf(rt); sRst[tid] = sinf(rt); sRp[tid] = ref_phi[tid];
  }
  const float cb = cconst[b];
  __syncthreads();

  // ---------------- phase 2: features ----------------
  if (tid < MT) {
    const int r = tid;
    const int gi = row0 + r;
    const float theta = coords[gi * 2 + 0];
    const float phi   = coords[gi * 2 + 1];
    const float ct = cosf(theta), st = sinf(theta);
#pragma unroll
    for (int j = 0; j < NREFS; ++j) {
      float cosd = ct * sRct[j] + st * sRst[j] * cosf(phi - sRp[j]);
      cosd = fminf(1.0f, fmaxf(-1.0f, cosd));
      sA[r * LDA + j] = (_Float16)(acosf(cosd) * (1.0f / PI_F));
    }
    sA[r * LDA + 10] = (_Float16)(theta * (1.0f / PI_F));
    sA[r * LDA + 11] = (_Float16)(phi * (1.0f / (2.0f * PI_F)));
    sA[r * LDA + 12] = (_Float16)1.0f;   // 1/R^2 feature
    sA[r * LDA + 13] = (_Float16)1.0f;   // bias column (weights row = b0)
  } else {
    const int r = tid - MT;
    for (int c = 14; c < 32; ++c) sA[r * LDA + c] = (_Float16)0.0f;
  }
  __syncthreads();

  const int myrow = wave * 16 + ln;     // this lane's data row

  // ---------------- phase 3: layer 0 (K=32, 1 WMMA per neuron tile) -------
  v8f acc[8];
  {
    const v16h bfeat = ld_bfrag(sA, myrow, LDA, 0, lh);
#pragma unroll
    for (int mt = 0; mt < 8; ++mt) {
      v16h aw = ld_afrag(sW0, mt * 16 + ln, LDA, 0, lh);
      v8f z;
#pragma unroll
      for (int i = 0; i < 8; ++i) z[i] = 0.0f;
      acc[mt] = wmma_f16(aw, bfeat, z);
    }
  }

  // ---- epilogue: LN (per-lane stats + one shuffle) + relu -> sH ----
  {
    float rs = 0.0f, rq = 0.0f;
#pragma unroll
    for (int mt = 0; mt < 8; ++mt)
#pragma unroll
      for (int i = 0; i < 8; ++i) { float x = acc[mt][i]; rs += x; rq += x * x; }
    rs += __shfl_xor(rs, 16, 32);
    rq += __shfl_xor(rq, 16, 32);
    const float mu  = rs * (1.0f / HID);
    const float var = rq * (1.0f / HID) - mu * mu;
    const float inv = rsqrtf(var + 1e-5f);
#pragma unroll
    for (int mt = 0; mt < 8; ++mt) {
      const v8f gv  = *(const v8f*)(sG0  + mt * 16 + lh * 8);
      const v8f bev = *(const v8f*)(sBe0 + mt * 16 + lh * 8);
      v8h hv;
#pragma unroll
      for (int i = 0; i < 8; ++i) {
        float y = (acc[mt][i] - mu) * inv * gv[i] + bev[i];
        hv[i] = (_Float16)fmaxf(y, 0.0f);
      }
      *(v8h*)(sH + myrow * LDH + mt * 16 + lh * 8) = hv;
    }
  }
  __syncthreads();

  // ---------------- phase 4: layer 1 (K=128, 4 WMMA per neuron tile) ------
  {
    v16h bh[4];
#pragma unroll
    for (int kc = 0; kc < 4; ++kc)
      bh[kc] = ld_bfrag(sH, myrow, LDH, kc * 32, lh);

#pragma unroll
    for (int mt = 0; mt < 8; ++mt) {
      v8f c;
#pragma unroll
      for (int i = 0; i < 8; ++i) c[i] = 0.0f;
#pragma unroll
      for (int kc = 0; kc < 4; ++kc) {
        v16h aw = ld_afrag(sW1, mt * 16 + ln, LDH, kc * 32, lh);
        c = wmma_f16(aw, bh[kc], c);
      }
      acc[mt] = c;
    }
  }

  // ---- epilogue: bias + LN + relu + fused combine-dot, write out ----
  {
    float rs = 0.0f, rq = 0.0f;
#pragma unroll
    for (int mt = 0; mt < 8; ++mt) {
      const v8f bv = *(const v8f*)(sB1 + mt * 16 + lh * 8);
#pragma unroll
      for (int i = 0; i < 8; ++i) {
        float x = acc[mt][i] + bv[i];
        acc[mt][i] = x; rs += x; rq += x * x;
      }
    }
    rs += __shfl_xor(rs, 16, 32);
    rq += __shfl_xor(rq, 16, 32);
    const float mu  = rs * (1.0f / HID);
    const float var = rq * (1.0f / HID) - mu * mu;
    const float inv = rsqrtf(var + 1e-5f);

    float op = 0.0f;
#pragma unroll
    for (int mt = 0; mt < 8; ++mt) {
      const v8f gv  = *(const v8f*)(sG1  + mt * 16 + lh * 8);
      const v8f bev = *(const v8f*)(sBe1 + mt * 16 + lh * 8);
      const v8f wcv = *(const v8f*)(sWc  + mt * 16 + lh * 8);
#pragma unroll
      for (int i = 0; i < 8; ++i) {
        float y = (acc[mt][i] - mu) * inv * gv[i] + bev[i];
        op += fmaxf(y, 0.0f) * wcv[i];
      }
    }
    op += __shfl_xor(op, 16, 32);
    if (lh == 0) out[row0 + wave * 16 + ln] = op + cb;
  }
}

// =====================  launch  =====================
extern "C" void kernel_launch(void* const* d_in, const int* in_sizes, int n_in,
                              void* d_out, int out_size, void* d_ws, size_t ws_size,
                              hipStream_t stream) {
  const float* coeffs = (const float*)d_in[0];
  const float* coords = (const float*)d_in[1];
  const float* cmean  = (const float*)d_in[2];
  const float* cstd   = (const float*)d_in[3];
  const float* rth    = (const float*)d_in[4];
  const float* rph    = (const float*)d_in[5];
  const float* b_w0 = (const float*)d_in[6];  const float* b_b0 = (const float*)d_in[7];
  const float* b_g0 = (const float*)d_in[8];  const float* b_be0 = (const float*)d_in[9];
  const float* b_w1 = (const float*)d_in[10]; const float* b_b1 = (const float*)d_in[11];
  const float* b_g1 = (const float*)d_in[12]; const float* b_be1 = (const float*)d_in[13];
  const float* b_w2 = (const float*)d_in[14]; const float* b_b2 = (const float*)d_in[15];
  const float* b_g2 = (const float*)d_in[16]; const float* b_be2 = (const float*)d_in[17];
  const float* b_wo = (const float*)d_in[18]; const float* b_bo = (const float*)d_in[19];
  const float* t_w0 = (const float*)d_in[20]; const float* t_b0 = (const float*)d_in[21];
  const float* t_g0 = (const float*)d_in[22]; const float* t_be0 = (const float*)d_in[23];
  const float* t_w1 = (const float*)d_in[24]; const float* t_b1 = (const float*)d_in[25];
  const float* t_g1 = (const float*)d_in[26]; const float* t_be1 = (const float*)d_in[27];
  const float* t_wo = (const float*)d_in[28]; const float* t_bo = (const float*)d_in[29];

  float* out = (float*)d_out;
  char* ws = (char*)d_ws;
  _Float16* hW0   = (_Float16*)(ws + WS_HW0);
  _Float16* hW1   = (_Float16*)(ws + WS_HW1);
  float*    wcomb = (float*)(ws + WS_WCOMB);
  float*    cconst= (float*)(ws + WS_CC);

  prep_kernel<<<88, 256, 0, stream>>>(t_w0, t_b0, t_w1, hW0, hW1);

  branch_kernel<<<BATCH, 128, 0, stream>>>(
      coeffs, cmean, cstd,
      b_w0, b_b0, b_g0, b_be0, b_w1, b_b1, b_g1, b_be1,
      b_w2, b_b2, b_g2, b_be2, b_wo, b_bo,
      t_wo, t_bo, wcomb, cconst);

  const int nblocks = (BATCH * NPB) / MT;       // 4096
  trunk_kernel<<<nblocks, 256, 0, stream>>>(
      coords, rth, rph, hW0, hW1,
      t_g0, t_be0, t_b1, t_g1, t_be1,
      wcomb, cconst, out);
}
